// CrossAttention_8976481649126
// MI455X (gfx1250) — matrix operand order repaired and checked by reading, not compiled, not run on hardware
//
#include <hip/hip_runtime.h>
#include <hip/hip_bf16.h>

typedef __bf16 bf16_t;
typedef __attribute__((ext_vector_type(16))) __bf16 v16bf;
typedef __attribute__((ext_vector_type(8)))  __bf16 v8bf;
typedef __attribute__((ext_vector_type(8)))  float  v8f;
typedef __attribute__((ext_vector_type(4)))  float  v4f;

#define C_DIM 1024
#define HEADS 16
#define HD    64
#define NQ    512
#define NKEY  2048
#define BATCH 4

#if defined(__gfx1250__) &&                                         \
    __has_builtin(__builtin_amdgcn_global_load_async_to_lds_b128) && \
    __has_builtin(__builtin_amdgcn_s_wait_asynccnt)
#define USE_ASYNC_LDS 1
#else
#define USE_ASYNC_LDS 0
#endif

static __device__ __forceinline__ v8f wmma_bf16(v16bf a, v16bf b, v8f c) {
  // D = A(16x32 bf16) * B(32x16 bf16) + C(16x16 f32)
  return __builtin_amdgcn_wmma_f32_16x16x32_bf16(false, a, false, b, (short)0, c,
                                                 false, false);
}

// -------------------------------------------------------------------------
// elementwise f32 -> bf16 (8 elems/thread, vectorized)
// -------------------------------------------------------------------------
__global__ __launch_bounds__(256)
void cvt_f32_bf16(const float* __restrict__ src, bf16_t* __restrict__ dst, int n8) {
  const int idx = blockIdx.x * 256 + threadIdx.x;
  if (idx < n8) {
    v4f a = *(const v4f*)(src + (size_t)idx * 8);
    v4f b = *(const v4f*)(src + (size_t)idx * 8 + 4);
    v8bf o;
#pragma unroll
    for (int i = 0; i < 4; ++i) { o[i] = (bf16_t)a[i]; o[4 + i] = (bf16_t)b[i]; }
    *(v8bf*)(dst + (size_t)idx * 8) = o;
  }
}

// -------------------------------------------------------------------------
// Y[m,n] = scale * sum_k X[m,k] * W[n,k]      X,W,Y: bf16  (W row-major [n,k])
// grid = (M/64, 1024/64), block = 128 (4 waves, one 16x64 tile each)
// -------------------------------------------------------------------------
__global__ __launch_bounds__(128)
void gemm_bf16_bf16(const bf16_t* __restrict__ X, const bf16_t* __restrict__ W,
                    bf16_t* __restrict__ Y, float scale) {
  const int lane = threadIdx.x & 31;
  const int w    = threadIdx.x >> 5;
  const int half = lane >> 4;
  const int ln   = lane & 15;
  const int mBase = blockIdx.x * 64 + w * 16;
  const int nBase = blockIdx.y * 64;

  v8f acc[4] = {};
  const bf16_t* xrow = X + (size_t)(mBase + ln) * C_DIM;

  for (int kc = 0; kc < C_DIM; kc += 32) {
    // A frag: elem i<8 -> k = half*8 + i ; i>=8 -> k = half*8 + 16 + (i-8)
    v16bf a;
    {
      const bf16_t* xa = xrow + kc + half * 8;
      v8bf g0 = *(const v8bf*)(xa);
      v8bf g1 = *(const v8bf*)(xa + 16);
#pragma unroll
      for (int i = 0; i < 8; ++i) { a[i] = g0[i]; a[8 + i] = g1[i]; }
    }
    // B frag: B[k][n] = W[n][k]; elem i<8 -> k = half*8 + i ; i>=8 -> k = half*8 + 16 + (i-8)
#pragma unroll
    for (int t = 0; t < 4; ++t) {
      const bf16_t* wr = W + (size_t)(nBase + t * 16 + ln) * C_DIM + kc + half * 8;
      v8bf g0 = *(const v8bf*)(wr);
      v8bf g1 = *(const v8bf*)(wr + 16);
      v16bf bb;
#pragma unroll
      for (int i = 0; i < 8; ++i) { bb[i] = g0[i]; bb[8 + i] = g1[i]; }
      acc[t] = wmma_bf16(a, bb, acc[t]);
    }
  }
#pragma unroll
  for (int t = 0; t < 4; ++t)
#pragma unroll
    for (int i = 0; i < 8; ++i)
      Y[(size_t)(mBase + i + half * 8) * C_DIM + nBase + t * 16 + ln] =
          (bf16_t)(acc[t][i] * scale);
}

// -------------------------------------------------------------------------
// out[m,n] = sum_k X[m,k]*W[n,k] + bias[n]    X,W: bf16, out: f32
// -------------------------------------------------------------------------
__global__ __launch_bounds__(128)
void gemm_bf16_f32_bias(const bf16_t* __restrict__ X, const bf16_t* __restrict__ W,
                        const float* __restrict__ bias, float* __restrict__ out) {
  const int lane = threadIdx.x & 31;
  const int w    = threadIdx.x >> 5;
  const int half = lane >> 4;
  const int ln   = lane & 15;
  const int mBase = blockIdx.x * 64 + w * 16;
  const int nBase = blockIdx.y * 64;

  v8f acc[4] = {};
  const bf16_t* xrow = X + (size_t)(mBase + ln) * C_DIM;

  for (int kc = 0; kc < C_DIM; kc += 32) {
    v16bf a;
    {
      const bf16_t* xa = xrow + kc + half * 8;
      v8bf g0 = *(const v8bf*)(xa);
      v8bf g1 = *(const v8bf*)(xa + 16);
#pragma unroll
      for (int i = 0; i < 8; ++i) { a[i] = g0[i]; a[8 + i] = g1[i]; }
    }
#pragma unroll
    for (int t = 0; t < 4; ++t) {
      const bf16_t* wr = W + (size_t)(nBase + t * 16 + ln) * C_DIM + kc + half * 8;
      v8bf g0 = *(const v8bf*)(wr);
      v8bf g1 = *(const v8bf*)(wr + 16);
      v16bf bb;
#pragma unroll
      for (int i = 0; i < 8; ++i) { bb[i] = g0[i]; bb[8 + i] = g1[i]; }
      acc[t] = wmma_bf16(a, bb, acc[t]);
    }
  }
#pragma unroll
  for (int t = 0; t < 4; ++t)
#pragma unroll
    for (int i = 0; i < 8; ++i) {
      const int n = nBase + t * 16 + ln;
      out[(size_t)(mBase + i + half * 8) * C_DIM + n] = acc[t][i] + bias[n];
    }
}

// -------------------------------------------------------------------------
// async staging of one 32-key x 64-dim K and V chunk slice
// (each of the 128 threads moves 16B of K and 16B of V, twice)
// builtin signature (from clang diagnostic): (AS1 int4*, AS3 int4*, imm, imm)
// -------------------------------------------------------------------------
#if USE_ASYNC_LDS
typedef __attribute__((ext_vector_type(4))) int v4i_t;
typedef __attribute__((address_space(1))) v4i_t* gv4i_p;
typedef __attribute__((address_space(3))) v4i_t* lv4i_p;
#endif

static __device__ __forceinline__ void stage_async(const bf16_t* kg, const bf16_t* vg,
                                                   bf16_t* kd, bf16_t* vd) {
#if USE_ASYNC_LDS
  __builtin_amdgcn_global_load_async_to_lds_b128((gv4i_p)(kg),     (lv4i_p)(kd),     0, 0);
  __builtin_amdgcn_global_load_async_to_lds_b128((gv4i_p)(kg + 8), (lv4i_p)(kd + 8), 0, 0);
  __builtin_amdgcn_global_load_async_to_lds_b128((gv4i_p)(vg),     (lv4i_p)(vd),     0, 0);
  __builtin_amdgcn_global_load_async_to_lds_b128((gv4i_p)(vg + 8), (lv4i_p)(vd + 8), 0, 0);
#endif
}

// -------------------------------------------------------------------------
// Flash attention: grid = (NQ/64, HEADS, BATCH), block = 128 (4 waves).
// Each wave owns 16 query rows; 32-key K/V chunks double-buffered in LDS.
// -------------------------------------------------------------------------
__global__ __launch_bounds__(128)
void attn_kernel(const bf16_t* __restrict__ qh, const bf16_t* __restrict__ kh,
                 const bf16_t* __restrict__ vh, const float* __restrict__ bias,
                 bf16_t* __restrict__ xh) {
  __shared__ bf16_t kbuf[2][32 * 64];
  __shared__ bf16_t vbuf[2][32 * 64];
  __shared__ bf16_t pbuf[4][16 * 32];

  const int tid  = threadIdx.x;
  const int lane = tid & 31;
  const int w    = tid >> 5;
  const int half = lane >> 4;
  const int ln   = lane & 15;
  const int qBase = blockIdx.x * 64 + w * 16;
  const int h = blockIdx.y;
  const int b = blockIdx.z;

  // Q A-fragments for the two 32-wide D chunks (loaded once)
  v16bf aQ[2];
  {
    const bf16_t* qrow = qh + ((size_t)(b * NQ + qBase + ln) * C_DIM) + h * HD;
#pragma unroll
    for (int dc = 0; dc < 2; ++dc) {
      const bf16_t* p = qrow + dc * 32 + half * 8;
      v8bf g0 = *(const v8bf*)(p);
      v8bf g1 = *(const v8bf*)(p + 16);
#pragma unroll
      for (int i = 0; i < 8; ++i) { aQ[dc][i] = g0[i]; aQ[dc][8 + i] = g1[i]; }
    }
  }

  float mrun[8], lrun[8];
#pragma unroll
  for (int i = 0; i < 8; ++i) { mrun[i] = -3.0e38f; lrun[i] = 0.0f; }
  v8f accO[4] = {};

  // staging map: thread -> (key_local = tid/4, 16-wide d slice)
  const int kl = tid >> 2;
  const int dL = (tid & 3) * 16;
  const size_t kvhead = (size_t)b * NKEY * C_DIM + (size_t)h * HD;
  const float* brow = bias + ((size_t)(b * HEADS + h) * NQ) * NKEY;

#if USE_ASYNC_LDS
  stage_async(kh + kvhead + (size_t)kl * C_DIM + dL,
              vh + kvhead + (size_t)kl * C_DIM + dL,
              &kbuf[0][kl * 64 + dL], &vbuf[0][kl * 64 + dL]);
#endif

  for (int kb = 0; kb < NKEY; kb += 32) {
    const int cur = (kb >> 5) & 1;
#if USE_ASYNC_LDS
    __builtin_amdgcn_s_wait_asynccnt(0);   // my chunk-cur copies have landed
    __syncthreads();                       // everyone's landed; prev buffer free
    if (kb + 32 < NKEY)
      stage_async(kh + kvhead + (size_t)(kb + 32 + kl) * C_DIM + dL,
                  vh + kvhead + (size_t)(kb + 32 + kl) * C_DIM + dL,
                  &kbuf[cur ^ 1][kl * 64 + dL], &vbuf[cur ^ 1][kl * 64 + dL]);
#else
    {
      const bf16_t* kg = kh + kvhead + (size_t)(kb + kl) * C_DIM + dL;
      const bf16_t* vg = vh + kvhead + (size_t)(kb + kl) * C_DIM + dL;
      *(v8bf*)&kbuf[cur][kl * 64 + dL]     = *(const v8bf*)(kg);
      *(v8bf*)&kbuf[cur][kl * 64 + dL + 8] = *(const v8bf*)(kg + 8);
      *(v8bf*)&vbuf[cur][kl * 64 + dL]     = *(const v8bf*)(vg);
      *(v8bf*)&vbuf[cur][kl * 64 + dL + 8] = *(const v8bf*)(vg + 8);
    }
    __syncthreads();
#endif

    // prefetch next chunk of the (huge, streamed-once) bias tensor
    if (kb + 32 < NKEY)
      __builtin_prefetch(&brow[(size_t)(qBase + ln) * NKEY + kb + 32], 0, 1);

    // scores S[16 rows x 32 keys] = Q · K^T  (two 16-key tiles, 2 WMMAs each)
    v8f s[2];
#pragma unroll
    for (int tk = 0; tk < 2; ++tk) {
      const int key_l = tk * 16 + ln;
      v8f z = {};
#pragma unroll
      for (int dc = 0; dc < 2; ++dc) {
        const bf16_t* kp = &kbuf[cur][key_l * 64 + dc * 32 + half * 8];
        v8bf g0 = *(const v8bf*)(kp);
        v8bf g1 = *(const v8bf*)(kp + 16);
        v16bf bb;
#pragma unroll
        for (int i = 0; i < 8; ++i) { bb[i] = g0[i]; bb[8 + i] = g1[i]; }
        z = wmma_bf16(aQ[dc], bb, z);
      }
      s[tk] = z;
    }

    // + bias (C-layout: elem i -> row qBase+i+8*half, col kb+tk*16+ln)
#pragma unroll
    for (int tk = 0; tk < 2; ++tk)
#pragma unroll
      for (int i = 0; i < 8; ++i)
        s[tk][i] += brow[(size_t)(qBase + i + half * 8) * NKEY + kb + tk * 16 + ln];

    // chunk row-max across the 16 lanes holding each row
    float cmax[8];
#pragma unroll
    for (int i = 0; i < 8; ++i) cmax[i] = fmaxf(s[0][i], s[1][i]);
#pragma unroll
    for (int off = 1; off < 16; off <<= 1)
#pragma unroll
      for (int i = 0; i < 8; ++i)
        cmax[i] = fmaxf(cmax[i], __shfl_xor(cmax[i], off, 32));

    float pr0[8], pr1[8], rsum[8];
#pragma unroll
    for (int i = 0; i < 8; ++i) {
      const float mn   = fmaxf(mrun[i], cmax[i]);
      const float corr = __expf(mrun[i] - mn);
      mrun[i] = mn;
      lrun[i] *= corr;
#pragma unroll
      for (int t = 0; t < 4; ++t) accO[t][i] *= corr;
      pr0[i] = __expf(s[0][i] - mn);
      pr1[i] = __expf(s[1][i] - mn);
      rsum[i] = pr0[i] + pr1[i];
    }
#pragma unroll
    for (int off = 1; off < 16; off <<= 1)
#pragma unroll
      for (int i = 0; i < 8; ++i) rsum[i] += __shfl_xor(rsum[i], off, 32);
#pragma unroll
    for (int i = 0; i < 8; ++i) lrun[i] += rsum[i];

    // re-layout P (C-layout -> A-layout) through per-wave LDS tile
#pragma unroll
    for (int i = 0; i < 8; ++i) {
      pbuf[w][(i + half * 8) * 32 + ln]      = (bf16_t)pr0[i];
      pbuf[w][(i + half * 8) * 32 + 16 + ln] = (bf16_t)pr1[i];
    }
    __builtin_amdgcn_wave_barrier();   // same-wave DS is in-order
    v16bf aP;
    {
      const bf16_t* pp = &pbuf[w][ln * 32 + half * 8];
      v8bf g0 = *(const v8bf*)(pp);
      v8bf g1 = *(const v8bf*)(pp + 16);
#pragma unroll
      for (int i = 0; i < 8; ++i) { aP[i] = g0[i]; aP[8 + i] = g1[i]; }
    }

    // O += P · V   (V B-frag: col = dim, elem i -> key_local = half*8 + (i&7) + 16*(i>>3))
#pragma unroll
    for (int t = 0; t < 4; ++t) {
      const int dim = t * 16 + ln;
      v16bf bv;
#pragma unroll
      for (int i = 0; i < 16; ++i) bv[i] = vbuf[cur][(half * 8 + (i & 7) + ((i >> 3) << 4)) * 64 + dim];
      accO[t] = wmma_bf16(aP, bv, accO[t]);
    }
  }

  // normalize and store x = O / l  as bf16 [B*NQ, C] at columns h*64..
#pragma unroll
  for (int t = 0; t < 4; ++t)
#pragma unroll
    for (int i = 0; i < 8; ++i) {
      const float o = accO[t][i] / lrun[i];
      xh[(size_t)(b * NQ + qBase + i + half * 8) * C_DIM + h * HD + t * 16 + ln] =
          (bf16_t)o;
    }
}

// -------------------------------------------------------------------------
extern "C" void kernel_launch(void* const* d_in, const int* in_sizes, int n_in,
                              void* d_out, int out_size, void* d_ws, size_t ws_size,
                              hipStream_t stream) {
  const float* q  = (const float*)d_in[0];
  const float* k  = (const float*)d_in[1];
  const float* v  = (const float*)d_in[2];
  const float* ab = (const float*)d_in[3];
  const float* wq = (const float*)d_in[4];
  const float* wk = (const float*)d_in[5];
  const float* wv = (const float*)d_in[6];
  const float* pw = (const float*)d_in[7];
  const float* pb = (const float*)d_in[8];
  float* out = (float*)d_out;

  char* ws = (char*)d_ws;
  const size_t MB = 1024 * 1024;
  bf16_t* qb  = (bf16_t*)(ws);            //  4 MiB  (q bf16)
  bf16_t* kbf = (bf16_t*)(ws + 4 * MB);   // 16 MiB  (k bf16)
  bf16_t* vbf = (bf16_t*)(ws + 20 * MB);  // 16 MiB  (v bf16)
  bf16_t* wqb = (bf16_t*)(ws + 36 * MB);  //  2 MiB
  bf16_t* wkb = (bf16_t*)(ws + 38 * MB);  //  2 MiB
  bf16_t* wvb = (bf16_t*)(ws + 40 * MB);  //  2 MiB
  bf16_t* pwb = (bf16_t*)(ws + 42 * MB);  //  2 MiB
  bf16_t* qh  = (bf16_t*)(ws + 44 * MB);  //  4 MiB  (Q proj)
  bf16_t* kh  = (bf16_t*)(ws + 48 * MB);  // 16 MiB  (K proj)
  bf16_t* vh  = (bf16_t*)(ws + 64 * MB);  // 16 MiB  (V proj)
  bf16_t* xh  = (bf16_t*)(ws + 80 * MB);  //  4 MiB  (attn out)

  const dim3 blk128(128);
  const dim3 blk256(256);
  const float qscale = 0.125f;   // 1/sqrt(64)

  // one-time f32 -> bf16 conversions (inputs + weights)
  cvt_f32_bf16<<<1024, blk256, 0, stream>>>(q,  qb,  (BATCH * NQ * C_DIM) / 8);
  cvt_f32_bf16<<<4096, blk256, 0, stream>>>(k,  kbf, (BATCH * NKEY * C_DIM) / 8);
  cvt_f32_bf16<<<4096, blk256, 0, stream>>>(v,  vbf, (BATCH * NKEY * C_DIM) / 8);
  cvt_f32_bf16<<<512,  blk256, 0, stream>>>(wq, wqb, (C_DIM * C_DIM) / 8);
  cvt_f32_bf16<<<512,  blk256, 0, stream>>>(wk, wkb, (C_DIM * C_DIM) / 8);
  cvt_f32_bf16<<<512,  blk256, 0, stream>>>(wv, wvb, (C_DIM * C_DIM) / 8);
  cvt_f32_bf16<<<512,  blk256, 0, stream>>>(pw, pwb, (C_DIM * C_DIM) / 8);

  // projections
  gemm_bf16_bf16<<<dim3(32, 16),  blk128, 0, stream>>>(qb,  wqb, qh, qscale);
  gemm_bf16_bf16<<<dim3(128, 16), blk128, 0, stream>>>(kbf, wkb, kh, 1.0f);
  gemm_bf16_bf16<<<dim3(128, 16), blk128, 0, stream>>>(vbf, wvb, vh, 1.0f);

  // fused flash attention with bias
  attn_kernel<<<dim3(NQ / 64, HEADS, BATCH), blk128, 0, stream>>>(qh, kh, vh, ab, xh);

  // output projection + bias -> f32
  gemm_bf16_f32_bias<<<dim3(32, 16), blk128, 0, stream>>>(xh, pwb, pb, out);
}
